// GraphTransformerv3_28329604284654
// MI455X (gfx1250) — compile-verified, matrix operand order checked by reading, not run on hardware
//
#include <hip/hip_runtime.h>
#include <math.h>

// ---------------- problem constants (from reference) ----------------
#define NN      16000      // total nodes (B*NPG)
#define EE      256000     // edges
#define HH      4          // heads
#define TEMB_H  64         // TEMB/2
#define OUTG    64         // gnn out = 2*C_OUT
#define COUT    32

// ---------------- WMMA types ----------------
typedef __attribute__((ext_vector_type(16))) __bf16 v16bf;
typedef __attribute__((ext_vector_type(8)))  float  v8f;

union ABFrag { v16bf v; __bf16 e[16]; };

// native f32 -> bf16 convert (backend emits v_cvt_*bf16* on gfx1250)
__device__ __forceinline__ __bf16 f2bf(float f) { return (__bf16)f; }

// ordered-uint encoding for float atomic max
__device__ __forceinline__ unsigned f2ord(float f) {
  unsigned u = __float_as_uint(f);
  return (u & 0x80000000u) ? ~u : (u | 0x80000000u);
}
__device__ __forceinline__ float ord2f(unsigned u) {
  return (u & 0x80000000u) ? __uint_as_float(u & 0x7FFFFFFFu) : __uint_as_float(~u);
}

// ---------------- bf16-WMMA GEMM, K and N known at compile time --------------
// C = [relu](A[M,K] @ W[K,N] + b).  Block = 128 threads = 4 waves; each wave
// computes one 16x16 tile of C; the 4 waves share one LDS-staged B tile.
template<int K, int N, bool RELU>
__global__ __launch_bounds__(128) void gemm_wmma(const float* __restrict__ A,
                                                 const float* __restrict__ Wm,
                                                 const float* __restrict__ bias,
                                                 float* __restrict__ C,
                                                 int M)
{
  __shared__ __bf16 Bt[16][32];      // transposed B tile: Bt[n][k]

  const int tid  = threadIdx.x;
  const int wave = tid >> 5;
  const int lane = tid & 31;
  const int m0 = (blockIdx.y * 4 + wave) * 16;
  const int n0 = blockIdx.x * 16;
  const bool active = (m0 < M);      // wave-uniform -> EXEC all-ones at WMMA

  const int r16 = lane & 15;         // row (A) / col (B/C) within tile
  const int kg  = lane >> 4;         // K-half selector
  const int col = n0 + r16;
  const float* Arow = A + (size_t)((active ? m0 : 0) + r16) * K;

  v8f acc = {};
#pragma unroll
  for (int k0 = 0; k0 < K; k0 += 32) {
    // ---- cooperative B tile load (coalesced along N), store transposed ----
    {
      const int r = tid >> 2;          // k within 32-block: 0..31
      const int q = (tid & 3) * 4;     // n offset: 0,4,8,12
      float4 wv = make_float4(0.f, 0.f, 0.f, 0.f);
      if (k0 + r < K)
        wv = *reinterpret_cast<const float4*>(Wm + (size_t)(k0 + r) * N + n0 + q);
      Bt[q + 0][r] = f2bf(wv.x);
      Bt[q + 1][r] = f2bf(wv.y);
      Bt[q + 2][r] = f2bf(wv.z);
      Bt[q + 3][r] = f2bf(wv.w);
    }
    __syncthreads();

    if (active) {
      ABFrag a, b;
      // ---- A fragment: two contiguous 8-float runs -> vector loads ----
#pragma unroll
      for (int run = 0; run < 2; ++run) {
        const int ks = k0 + run * 16 + kg * 8;
        if (ks + 8 <= K) {
          const float4 f0 = *reinterpret_cast<const float4*>(Arow + ks);
          const float4 f1 = *reinterpret_cast<const float4*>(Arow + ks + 4);
          a.e[run * 8 + 0] = f2bf(f0.x); a.e[run * 8 + 1] = f2bf(f0.y);
          a.e[run * 8 + 2] = f2bf(f0.z); a.e[run * 8 + 3] = f2bf(f0.w);
          a.e[run * 8 + 4] = f2bf(f1.x); a.e[run * 8 + 5] = f2bf(f1.y);
          a.e[run * 8 + 6] = f2bf(f1.z); a.e[run * 8 + 7] = f2bf(f1.w);
        } else {
#pragma unroll
          for (int j = 0; j < 8; ++j)
            a.e[run * 8 + j] = f2bf((ks + j) < K ? Arow[ks + j] : 0.0f);
        }
      }
      // ---- B fragment from LDS: two contiguous 16-byte runs per lane ----
#pragma unroll
      for (int run = 0; run < 2; ++run) {
        const int ks = run * 16 + kg * 8;
#pragma unroll
        for (int j = 0; j < 8; ++j)
          b.e[run * 8 + j] = Bt[r16][ks + j];
      }
      acc = __builtin_amdgcn_wmma_f32_16x16x32_bf16(
          /*neg_a=*/false, a.v, /*neg_b=*/false, b.v,
          /*c_mod=*/(short)0, acc, /*reuse_a=*/false, /*reuse_b=*/false);
    }
    __syncthreads();
  }

  if (active) {
    const float bv = bias ? bias[col] : 0.0f;
    float* Cbase = C + (size_t)(m0 + 8 * kg) * N + col;   // constant stride N
#pragma unroll
    for (int r = 0; r < 8; ++r) {
      float v = acc[r] + bv;
      if (RELU) v = fmaxf(v, 0.0f);
      Cbase[r * N] = v;                                   // immediate offsets
    }
  }
}

// ---------------- time embedding: te[n,j] = b[j] + sum_i sin/cos(t*f_i)*W[i,j]
__global__ void time_embed_kernel(const float* __restrict__ t,
                                  const float* __restrict__ Wt,
                                  const float* __restrict__ bt,
                                  float* __restrict__ te, int n_nodes)
{
  const int idx = blockIdx.x * blockDim.x + threadIdx.x;
  if (idx >= n_nodes * OUTG) return;
  const int n = idx / OUTG, j = idx % OUTG;
  const float tv = t[n];
  const float lf = -logf(10000.0f) / (float)(TEMB_H - 1);
  float acc = bt[j];
  for (int i = 0; i < TEMB_H; ++i) {
    const float ph = tv * __expf(lf * (float)i);
    acc += __sinf(ph) * Wt[(size_t)i * OUTG + j]
         + __cosf(ph) * Wt[(size_t)(TEMB_H + i) * OUTG + j];
  }
  te[idx] = acc;
}

// ---------------- xfc = relu(h) * te[:,2c] + te[:,2c+1] ----------------
__global__ void film_xfc_kernel(const float* __restrict__ h,
                                const float* __restrict__ te,
                                float* __restrict__ xfc, int n_nodes)
{
  const int idx = blockIdx.x * blockDim.x + threadIdx.x;
  if (idx >= n_nodes * COUT) return;
  const int n = idx / COUT, c = idx % COUT;
  const float v = fmaxf(h[idx], 0.0f);
  xfc[idx] = v * te[(size_t)n * OUTG + 2 * c] + te[(size_t)n * OUTG + 2 * c + 1];
}

// ---------------- fill scratch with a u32 pattern ----------------
__global__ void fill_u32_kernel(unsigned* __restrict__ p, unsigned v, size_t n)
{
  const size_t i = (size_t)blockIdx.x * blockDim.x + threadIdx.x;
  if (i < n) p[i] = v;
}

// ---- edge pass 1: score + atomic-max per (dst,head) (ordered uint) ----
template<int C>
__global__ void edge_score_kernel(const float* __restrict__ q,
                                  const float* __restrict__ k,
                                  const float* __restrict__ eW,   // [H*C]
                                  const float* __restrict__ ew,
                                  const int* __restrict__ src,
                                  const int* __restrict__ dst,
                                  float* __restrict__ score,
                                  unsigned* __restrict__ mEnc)
{
  const int idx = blockIdx.x * blockDim.x + threadIdx.x;
  if (idx >= EE * HH) return;
  const int e = idx >> 2, h = idx & 3;        // HH == 4
  const int s = src[e], d = dst[e];
  const float w = ew[e];
  const float4* qp = (const float4*)(q + ((size_t)d * HH + h) * C);
  const float4* kp = (const float4*)(k + ((size_t)s * HH + h) * C);
  const float4* ep = (const float4*)(eW + (size_t)h * C);
  if (e + 1 < EE) __builtin_prefetch(k + ((size_t)src[e + 1] * HH + h) * C, 0, 1);
  float acc = 0.f;
#pragma unroll
  for (int c4 = 0; c4 < C / 4; ++c4) {
    const float4 qv = qp[c4], kv = kp[c4], ev = ep[c4];
    acc += qv.x * (kv.x + w * ev.x) + qv.y * (kv.y + w * ev.y)
         + qv.z * (kv.z + w * ev.z) + qv.w * (kv.w + w * ev.w);
  }
  acc *= rsqrtf((float)C);
  score[idx] = acc;
  atomicMax(mEnc + (size_t)d * HH + h, f2ord(acc));
}

// ---- edge pass 2: ex = exp(score - m); den += ex; num += ex*(v+e) ----
template<int C>
__global__ void edge_accum_kernel(const float* __restrict__ v,
                                  const float* __restrict__ eW,
                                  const float* __restrict__ ew,
                                  const int* __restrict__ src,
                                  const int* __restrict__ dst,
                                  const float* __restrict__ score,
                                  const unsigned* __restrict__ mEnc,
                                  float* __restrict__ den,
                                  float* __restrict__ num)
{
  const int idx = blockIdx.x * blockDim.x + threadIdx.x;
  if (idx >= EE * HH) return;
  const int e = idx >> 2, h = idx & 3;        // HH == 4
  const int s = src[e], d = dst[e];
  const float m = ord2f(mEnc[(size_t)d * HH + h]);
  const float ex = __expf(score[idx] - m);
  atomicAdd(den + (size_t)d * HH + h, ex);
  const float w = ew[e];
  const float4* vp = (const float4*)(v + ((size_t)s * HH + h) * C);
  const float4* ep = (const float4*)(eW + (size_t)h * C);
  float* np = num + ((size_t)d * HH + h) * C;
#pragma unroll
  for (int c4 = 0; c4 < C / 4; ++c4) {
    const float4 vv = vp[c4], ev = ep[c4];
    atomicAdd(np + 4 * c4 + 0, ex * (vv.x + w * ev.x));
    atomicAdd(np + 4 * c4 + 1, ex * (vv.y + w * ev.y));
    atomicAdd(np + 4 * c4 + 2, ex * (vv.z + w * ev.z));
    atomicAdd(np + 4 * c4 + 3, ex * (vv.w + w * ev.w));
  }
}

// ---- per-node finalize: g = mean_h(num/den) + skip, optional relu ----
template<int C>
__global__ void node_finalize_kernel(const float* __restrict__ num,
                                     const float* __restrict__ den,
                                     const float* __restrict__ skip,
                                     float* __restrict__ g,
                                     int relu)
{
  const int idx = blockIdx.x * blockDim.x + threadIdx.x;
  if (idx >= NN * C) return;
  const int n = idx / C, c = idx % C;
  float acc = 0.f;
#pragma unroll
  for (int h = 0; h < HH; ++h) {
    const float dd = den[(size_t)n * HH + h];
    const float nn = num[((size_t)n * HH + h) * C + c];
    acc += (dd > 0.f) ? nn / dd : 0.f;
  }
  acc = acc * (1.0f / HH) + skip[(size_t)n * C + c];
  if (relu) acc = fmaxf(acc, 0.f);
  g[idx] = acc;
}

// ---- training-mode BatchNorm1d: one block per channel ----
__global__ void batchnorm_kernel(float* __restrict__ g,
                                 const float* __restrict__ w,
                                 const float* __restrict__ b, int C)
{
  const int c = blockIdx.x;
  __shared__ float ssum[256], ssq[256];
  float s = 0.f, q = 0.f;
  for (int n = threadIdx.x; n < NN; n += blockDim.x) {
    const float v = g[(size_t)n * C + c];
    s += v; q += v * v;
  }
  ssum[threadIdx.x] = s; ssq[threadIdx.x] = q;
  __syncthreads();
  for (int o = blockDim.x / 2; o > 0; o >>= 1) {
    if (threadIdx.x < (unsigned)o) {
      ssum[threadIdx.x] += ssum[threadIdx.x + o];
      ssq[threadIdx.x]  += ssq[threadIdx.x + o];
    }
    __syncthreads();
  }
  const float mu  = ssum[0] / (float)NN;
  const float var = ssq[0] / (float)NN - mu * mu;
  const float sc  = w[c] * rsqrtf(var + 1e-5f);
  const float bi  = b[c] - mu * sc;
  for (int n = threadIdx.x; n < NN; n += blockDim.x) {
    const size_t i = (size_t)n * C + c;
    g[i] = g[i] * sc + bi;
  }
}

// ---- final FiLM: out = xfc * g[:,2c] + g[:,2c+1] ----
__global__ void film_out_kernel(const float* __restrict__ xfc,
                                const float* __restrict__ g,
                                float* __restrict__ out)
{
  const int idx = blockIdx.x * blockDim.x + threadIdx.x;
  if (idx >= NN * COUT) return;
  const int n = idx / COUT, c = idx % COUT;
  out[idx] = xfc[idx] * g[(size_t)n * OUTG + 2 * c]
           + g[(size_t)n * OUTG + 2 * c + 1];
}

// =====================================================================
extern "C" void kernel_launch(void* const* d_in, const int* in_sizes, int n_in,
                              void* d_out, int out_size, void* d_ws, size_t ws_size,
                              hipStream_t stream)
{
  (void)in_sizes; (void)n_in; (void)out_size; (void)ws_size;

  // ---- inputs (pytree-flattened, sorted-key order inside params) ----
  const float* x  = (const float*)d_in[0];
  const float* y  = (const float*)d_in[1];
  const float* t  = (const float*)d_in[2];
  const float* ew = (const float*)d_in[3];
  // params.bn: 3 x {b,w}
  const float* bnb[3]; const float* bnw[3];
  for (int i = 0; i < 3; ++i) { bnb[i] = (const float*)d_in[4 + 2*i]; bnw[i] = (const float*)d_in[5 + 2*i]; }
  // params.conv: 4 x {e.W, k.W, k.b, q.W, q.b, skip.W, skip.b, v.W, v.b}
  struct ConvP { const float *eW,*kW,*kb,*qW,*qb,*sW,*sb,*vW,*vb; } cv[4];
  for (int i = 0; i < 4; ++i) {
    const int b0 = 10 + 9*i;
    cv[i].eW=(const float*)d_in[b0+0]; cv[i].kW=(const float*)d_in[b0+1]; cv[i].kb=(const float*)d_in[b0+2];
    cv[i].qW=(const float*)d_in[b0+3]; cv[i].qb=(const float*)d_in[b0+4];
    cv[i].sW=(const float*)d_in[b0+5]; cv[i].sb=(const float*)d_in[b0+6];
    cv[i].vW=(const float*)d_in[b0+7]; cv[i].vb=(const float*)d_in[b0+8];
  }
  // params.mlp: 4 x {W,b}
  const float* mW[4]; const float* mb[4];
  for (int i = 0; i < 4; ++i) { mW[i] = (const float*)d_in[46 + 2*i]; mb[i] = (const float*)d_in[47 + 2*i]; }
  const float* tW = (const float*)d_in[54];
  const float* tb = (const float*)d_in[55];
  const int* ei = (const int*)d_in[56];
  const int* src = ei;
  const int* dst = ei + EE;

  // ---- workspace layout (mEnc/den/num contiguous for one fill) ----
  float* W = (float*)d_ws;
  float* te    = W; W += (size_t)NN * OUTG;
  float* h0    = W; W += (size_t)NN * 256;
  float* h1    = W; W += (size_t)NN * 256;
  float* xfc   = W; W += (size_t)NN * COUT;
  float* qb_   = W; W += (size_t)NN * 256;
  float* kb_   = W; W += (size_t)NN * 256;
  float* vb_   = W; W += (size_t)NN * 256;
  float* skip_ = W; W += (size_t)NN * OUTG;
  float* score = W; W += (size_t)EE * HH;
  unsigned* mEnc = (unsigned*)W; W += (size_t)NN * HH;
  float* den   = W; W += (size_t)NN * HH;
  float* num   = W; W += (size_t)NN * HH * 64;
  float* g0    = W; W += (size_t)NN * OUTG;
  float* g1    = W; W += (size_t)NN * OUTG;

  auto grid_for = [](int M, int N_) { return dim3(N_ / 16, (M / 16 + 3) / 4); };

  // ---- time embedding ----
  time_embed_kernel<<<(NN * OUTG + 255) / 256, 256, 0, stream>>>(t, tW, tb, te, NN);

  // ---- MLP (WMMA bf16), final relu fused into FiLM ----
  gemm_wmma<16,  256, true ><<<grid_for(NN, 256), 128, 0, stream>>>(x,  mW[0], mb[0], h0, NN);
  gemm_wmma<256, 256, true ><<<grid_for(NN, 256), 128, 0, stream>>>(h0, mW[1], mb[1], h1, NN);
  gemm_wmma<256, 256, true ><<<grid_for(NN, 256), 128, 0, stream>>>(h1, mW[2], mb[2], h0, NN);
  gemm_wmma<256, 32,  false><<<grid_for(NN, 32),  128, 0, stream>>>(h0, mW[3], mb[3], h1, NN);
  film_xfc_kernel<<<(NN * COUT + 255) / 256, 256, 0, stream>>>(h1, te, xfc, NN);

  // ---- GNN: 4 TransformerConv layers ----
  const float* gin = y;
  float* gbuf[2] = {g0, g1};
  int cur = 0;
  const int EHt = EE * HH;
  for (int L = 0; L < 4; ++L) {
    const int dout = (L == 3) ? 64 : 32;
    const int HC   = HH * dout;

    if (L == 0) {              // din=1, dout=32, HC=128
      gemm_wmma<1, 128, false><<<grid_for(NN, 128), 128, 0, stream>>>(gin, cv[L].qW, cv[L].qb, qb_,  NN);
      gemm_wmma<1, 128, false><<<grid_for(NN, 128), 128, 0, stream>>>(gin, cv[L].kW, cv[L].kb, kb_,  NN);
      gemm_wmma<1, 128, false><<<grid_for(NN, 128), 128, 0, stream>>>(gin, cv[L].vW, cv[L].vb, vb_,  NN);
      gemm_wmma<1, 32,  false><<<grid_for(NN, 32),  128, 0, stream>>>(gin, cv[L].sW, cv[L].sb, skip_, NN);
    } else if (L < 3) {        // din=32, dout=32, HC=128
      gemm_wmma<32, 128, false><<<grid_for(NN, 128), 128, 0, stream>>>(gin, cv[L].qW, cv[L].qb, qb_,  NN);
      gemm_wmma<32, 128, false><<<grid_for(NN, 128), 128, 0, stream>>>(gin, cv[L].kW, cv[L].kb, kb_,  NN);
      gemm_wmma<32, 128, false><<<grid_for(NN, 128), 128, 0, stream>>>(gin, cv[L].vW, cv[L].vb, vb_,  NN);
      gemm_wmma<32, 32,  false><<<grid_for(NN, 32),  128, 0, stream>>>(gin, cv[L].sW, cv[L].sb, skip_, NN);
    } else {                   // din=32, dout=64, HC=256
      gemm_wmma<32, 256, false><<<grid_for(NN, 256), 128, 0, stream>>>(gin, cv[L].qW, cv[L].qb, qb_,  NN);
      gemm_wmma<32, 256, false><<<grid_for(NN, 256), 128, 0, stream>>>(gin, cv[L].kW, cv[L].kb, kb_,  NN);
      gemm_wmma<32, 256, false><<<grid_for(NN, 256), 128, 0, stream>>>(gin, cv[L].vW, cv[L].vb, vb_,  NN);
      gemm_wmma<32, 64,  false><<<grid_for(NN, 64),  128, 0, stream>>>(gin, cv[L].sW, cv[L].sb, skip_, NN);
    }

    const size_t fillN = (size_t)NN * HH * (size_t)(2 + dout);  // mEnc+den+num
    fill_u32_kernel<<<(unsigned)((fillN + 255) / 256), 256, 0, stream>>>(mEnc, 0u, fillN);

    if (dout == 32) {
      edge_score_kernel<32><<<(EHt + 255) / 256, 256, 0, stream>>>(qb_, kb_, cv[L].eW, ew, src, dst, score, mEnc);
      edge_accum_kernel<32><<<(EHt + 255) / 256, 256, 0, stream>>>(vb_, cv[L].eW, ew, src, dst, score, mEnc, den, num);
      node_finalize_kernel<32><<<(NN * 32 + 255) / 256, 256, 0, stream>>>(num, den, skip_, gbuf[cur], (L < 3) ? 1 : 0);
    } else {
      edge_score_kernel<64><<<(EHt + 255) / 256, 256, 0, stream>>>(qb_, kb_, cv[L].eW, ew, src, dst, score, mEnc);
      edge_accum_kernel<64><<<(EHt + 255) / 256, 256, 0, stream>>>(vb_, cv[L].eW, ew, src, dst, score, mEnc, den, num);
      node_finalize_kernel<64><<<(NN * 64 + 255) / 256, 256, 0, stream>>>(num, den, skip_, gbuf[cur], (L < 3) ? 1 : 0);
    }
    if (L < 3)
      batchnorm_kernel<<<32, 256, 0, stream>>>(gbuf[cur], bnw[L], bnb[L], 32);

    gin = gbuf[cur];
    cur ^= 1;
  }

  // ---- final FiLM to output ----
  film_out_kernel<<<(NN * COUT + 255) / 256, 256, 0, stream>>>(xfc, gin, (float*)d_out);
}
